// EvaAttention_73761768341885
// MI455X (gfx1250) — matrix operand order repaired and checked by reading, not compiled, not run on hardware
//
#include <hip/hip_runtime.h>
#include <hip/hip_bf16.h>

typedef __attribute__((ext_vector_type(16))) _Float16 v16h;
typedef __attribute__((ext_vector_type(8)))  _Float16 v8h;
typedef __attribute__((ext_vector_type(8)))  float    v8f;
typedef __attribute__((ext_vector_type(4)))  int      v4i;

#define BB 16
#define NN 1025
#define CC 1024
#define HH 16
#define DD 64
#define VSTR 1040          // padded Vt row stride (multiple of 16 halves)
#define MTOT (BB*NN)       // 16400
#define SCALEQ 0.125f      // D^-0.5
#define EPSF 1e-6f

__device__ inline v16h cat16(v8h a, v8h b){
  v16h r;
#pragma unroll
  for (int i=0;i<8;i++){ r[i]=a[i]; r[i+8]=b[i]; }
  return r;
}

__device__ inline v8f wmma_f16(v16h a, v16h b, v8f c){
  return __builtin_amdgcn_wmma_f32_16x16x32_f16(false, a, false, b, (short)0, c, false, false);
}

__device__ inline float wave_sum32(float v){
  v += __shfl_xor(v,1,32); v += __shfl_xor(v,2,32);
  v += __shfl_xor(v,4,32); v += __shfl_xor(v,8,32);
  v += __shfl_xor(v,16,32);
  return v;
}

// ---- CDNA5 async global->LDS staging (ASYNCcnt path), with safe fallback ----
#if __has_builtin(__builtin_amdgcn_global_load_async_to_lds_b128)
#define HAVE_ASYNC_LDS 1
typedef __attribute__((address_space(1))) v4i gv4i_t;   // global v4i
typedef __attribute__((address_space(3))) v4i lv4i_t;   // LDS v4i
__device__ inline void async_copy16h(const _Float16* g, _Float16* l){
  // per-lane: 16 bytes global -> LDS, tracked by ASYNCcnt
  __builtin_amdgcn_global_load_async_to_lds_b128(
      (gv4i_t*)(unsigned long long)(uintptr_t)g,
      (lv4i_t*)(unsigned int)(uintptr_t)l, 0, 0);
}
__device__ inline void wait_async0(){
#if __has_builtin(__builtin_amdgcn_s_wait_asynccnt)
  __builtin_amdgcn_s_wait_asynccnt(0);
#else
  asm volatile("s_wait_asynccnt 0x0" ::: "memory");
#endif
}
#else
#define HAVE_ASYNC_LDS 0
#endif

// ---------------- elementwise helpers ----------------
__global__ __launch_bounds__(256) void f32_to_f16_kernel(const float* __restrict__ in,
                                                         _Float16* __restrict__ out, long n){
  long i = (long)blockIdx.x*256 + threadIdx.x;
  if (i < n) out[i] = (_Float16)in[i];
}

__global__ __launch_bounds__(256) void bias_cat_kernel(const float* __restrict__ qb,
                                                       const float* __restrict__ kb,
                                                       const float* __restrict__ vb,
                                                       float* __restrict__ out){
  int i = blockIdx.x*256 + threadIdx.x;
  if (i < 1024)       out[i] = qb[i];
  else if (i < 2048)  out[i] = kb[i-1024];
  else if (i < 3072)  out[i] = vb[i-2048];
}

// ---------------- GEMM: out[M,N] = A[M,K] * W[N,K]^T + bias ----------------
// 256 threads (8 waves), 128x256 block tile, each wave a 64x64 sub-tile
// (4x4 WMMA tiles -> 16 WMMAs per 8 fragment loads), K-step 32.
__global__ __launch_bounds__(256) void gemm_f16_kernel(const _Float16* __restrict__ A,
                                                       const _Float16* __restrict__ W,
                                                       const float* __restrict__ bias,
                                                       float* __restrict__ C,
                                                       int M, int N, int K){
  __shared__ __align__(64) _Float16 sA[128*32];   // 8 KB
  __shared__ __align__(64) _Float16 sB[256*32];   // 16 KB
  const int tid  = threadIdx.x;
  const int lane = tid & 31, wave = tid >> 5;
  const int l15  = lane & 15;
  const int half = lane >> 4;                 // 0: lanes 0-15, 1: lanes 16-31
  const int mtile = blockIdx.x * 128, ntile = blockIdx.y * 256;
  const int wm = (wave & 1) * 64;             // 2 waves along M
  const int wn = (wave >> 1) * 64;            // 4 waves along N

  v8f acc[4][4] = {};

  const int arow  = tid >> 1;   const int akoff = (tid & 1) * 16;   // 128x32: 16 halves/thread
  const int brow  = tid;                                            // 256x32: 32 halves/thread
  int a_grow = mtile + arow; if (a_grow > M-1) a_grow = M-1;
  const size_t a_goff = (size_t)a_grow*K + akoff;
  const size_t b_goff = (size_t)(ntile + brow)*K;

  for (int k0 = 0; k0 < K; k0 += 32){
#if HAVE_ASYNC_LDS
    async_copy16h(&A[a_goff + k0],     &sA[arow*32 + akoff]);
    async_copy16h(&A[a_goff + k0 + 8], &sA[arow*32 + akoff + 8]);
    async_copy16h(&W[b_goff + k0],      &sB[brow*32]);
    async_copy16h(&W[b_goff + k0 + 8],  &sB[brow*32 + 8]);
    async_copy16h(&W[b_goff + k0 + 16], &sB[brow*32 + 16]);
    async_copy16h(&W[b_goff + k0 + 24], &sB[brow*32 + 24]);
    wait_async0();
#else
    *(v16h*)&sA[arow*32 + akoff] = *(const v16h*)&A[a_goff + k0];
    *(v16h*)&sB[brow*32]         = *(const v16h*)&W[b_goff + k0];
    *(v16h*)&sB[brow*32 + 16]    = *(const v16h*)&W[b_goff + k0 + 16];
#endif
    __syncthreads();

    v16h af[4], bf[4];
    const int ako = half ? 16 : 0;            // A layout: lanes<16 K 0..15, lanes>=16 K 16..31
#pragma unroll
    for (int i=0;i<4;i++){
      int r = wm + i*16 + l15;
      v8h lo = *(const v8h*)&sA[r*32 + ako];
      v8h hi = *(const v8h*)&sA[r*32 + ako + 8];
      af[i] = cat16(lo, hi);
    }
#pragma unroll
    for (int j=0;j<4;j++){
      int cidx = wn + j*16 + l15;             // B layout: lanes<16 K 0..15, lanes>=16 K 16..31
      bf[j] = *(const v16h*)&sB[cidx*32 + (half ? 16 : 0)];
    }
#pragma unroll
    for (int i=0;i<4;i++)
#pragma unroll
      for (int j=0;j<4;j++)
        acc[i][j] = wmma_f16(af[i], bf[j], acc[i][j]);
    __syncthreads();
  }

#pragma unroll
  for (int i=0;i<4;i++)
#pragma unroll
    for (int j=0;j<4;j++){
      int n = ntile + wn + j*16 + l15;
      float bv = bias ? bias[n] : 0.f;
#pragma unroll
      for (int r=0;r<8;r++){
        int m = mtile + wm + i*16 + r + half*8;   // D layout: VGPR r -> row r (+8 for upper half)
        if (m < M) C[(size_t)m*N + n] = acc[i][j][r] + bv;
      }
    }
}

// ---------------- QK layernorm + RoPE + pack to f16 (and V transpose) ----------------
// one wave per (b,n,h); lane owns d pair (2*lane, 2*lane+1) -> rope pair stays in-lane.
__global__ __launch_bounds__(256) void qk_prep_kernel(const float* __restrict__ qkv,
                                                      const float* __restrict__ rope,
                                                      const float* __restrict__ qn_g, const float* __restrict__ qn_b,
                                                      const float* __restrict__ kn_g, const float* __restrict__ kn_b,
                                                      _Float16* __restrict__ qh,
                                                      _Float16* __restrict__ kh,
                                                      _Float16* __restrict__ vT){
  const int lane = threadIdx.x & 31, wave = threadIdx.x >> 5;
  long row = (long)blockIdx.x*8 + wave;                 // over B*N*H rows: ((b*N+n)*H + h)
  if (row >= (long)BB*NN*HH) return;                    // uniform per wave
  const int h = (int)(row % HH);
  const long bn = row / HH;
  const int n = (int)(bn % NN);
  const int b = (int)(bn / NN);
  const float* base = qkv + bn*(3*CC) + h*DD;           // q at +0, k at +CC, v at +2*CC
  const int d0 = lane*2;

  float s0=0.f, s1=0.f, c0=0.f, c1=0.f;
  if (n > 0){
    const float* rr = rope + (size_t)(n-1)*(2*DD);
    s0 = rr[d0]; s1 = rr[d0+1]; c0 = rr[DD+d0]; c1 = rr[DD+d0+1];
  }

#pragma unroll
  for (int w=0; w<2; w++){
    const float* src = base + w*CC;
    float x0 = src[d0], x1 = src[d0+1];
    float mu  = wave_sum32(x0 + x1) * (1.f/64.f);
    float dx0 = x0 - mu, dx1 = x1 - mu;
    float var = wave_sum32(dx0*dx0 + dx1*dx1) * (1.f/64.f);
    float rs  = rsqrtf(var + EPSF);
    const float* g  = w ? kn_g : qn_g;
    const float* bb = w ? kn_b : qn_b;
    float y0 = dx0*rs*g[d0]   + bb[d0];
    float y1 = dx1*rs*g[d0+1] + bb[d0+1];
    if (n > 0){
      float t0 = y0*c0 + (-y1)*s0;    // rot: out[2i]=-x[2i+1], out[2i+1]=x[2i]
      float t1 = y1*c1 + ( y0)*s1;
      y0 = t0; y1 = t1;
    }
    if (w == 0){ y0 *= SCALEQ; y1 *= SCALEQ; }
    _Float16* dst = (w ? kh : qh) + ((size_t)(b*HH + h)*NN + n)*DD + d0;
    dst[0] = (_Float16)y0; dst[1] = (_Float16)y1;
  }
  // V -> transposed f16 (B,H,D,VSTR)
  const float* vsrc = base + 2*CC;
  size_t vo = ((size_t)(b*HH + h)*DD + d0)*VSTR + n;
  vT[vo]        = (_Float16)vsrc[d0];
  vT[vo + VSTR] = (_Float16)vsrc[d0+1];
}

// ---------------- flash attention ----------------
// grid (qtiles=17, bh=256), 128 threads (4 waves), each wave: 16 query rows.
__global__ __launch_bounds__(128) void flash_kernel(const _Float16* __restrict__ q,
                                                    const _Float16* __restrict__ k,
                                                    const _Float16* __restrict__ vT,
                                                    float* __restrict__ out){
  __shared__ __align__(64) _Float16 p_lds[4][16*64];
  const int bh = blockIdx.y;
  const int b = bh >> 4, h = bh & 15;
  const int wave = threadIdx.x >> 5, lane = threadIdx.x & 31;
  const int l15 = lane & 15, half = lane >> 4;
  const int q0 = blockIdx.x*64 + wave*16;

  const _Float16* qp = q  + (size_t)bh * NN * DD;
  const _Float16* kp = k  + (size_t)bh * NN * DD;
  const _Float16* vp = vT + (size_t)bh * DD * VSTR;

  // Q fragments (A-matrix layout), 2 k-steps of 32 over D=64
  v16h aq[2];
  {
    int r = q0 + l15; if (r > NN-1) r = NN-1;
    const int ako = half ? 16 : 0;
#pragma unroll
    for (int s=0;s<2;s++){
      v8h lo = *(const v8h*)&qp[(size_t)r*DD + 32*s + ako];
      v8h hi = *(const v8h*)&qp[(size_t)r*DD + 32*s + ako + 8];
      aq[s] = cat16(lo, hi);
    }
  }

  v8f oacc[4] = {};
  float mrow[8], lrow[8];
#pragma unroll
  for (int r=0;r<8;r++){ mrow[r] = -1e30f; lrow[r] = 0.f; }

  for (int kb = 0; kb < NN; kb += 64){
    // ---- S = Q * K^T over this KV tile (4 x 16 keys) ----
    v8f s_acc[4] = {};
#pragma unroll
    for (int j=0;j<4;j++){
      int key = kb + j*16 + l15; if (key > NN-1) key = NN-1;
#pragma unroll
      for (int s=0;s<2;s++){
        v16h bk = *(const v16h*)&kp[(size_t)key*DD + 32*s + (half ? 16 : 0)];
        s_acc[j] = wmma_f16(aq[s], bk, s_acc[j]);
      }
      if (kb + j*16 + l15 >= NN){
#pragma unroll
        for (int r=0;r<8;r++) s_acc[j][r] = -1e30f;
      }
    }
    // ---- online softmax: stats per row r (row = r + 8*half), reduce within 16-lane half ----
    float nm[8], fct[8];
#pragma unroll
    for (int r=0;r<8;r++){
      float t = fmaxf(fmaxf(s_acc[0][r], s_acc[1][r]), fmaxf(s_acc[2][r], s_acc[3][r]));
      t = fmaxf(t, __shfl_xor(t,1,32));
      t = fmaxf(t, __shfl_xor(t,2,32));
      t = fmaxf(t, __shfl_xor(t,4,32));
      t = fmaxf(t, __shfl_xor(t,8,32));
      nm[r]  = fmaxf(mrow[r], t);
      fct[r] = __expf(mrow[r] - nm[r]);
      mrow[r] = nm[r];
    }
#pragma unroll
    for (int r=0;r<8;r++){
      const int rowl = r + half*8;
      float rs = 0.f;
#pragma unroll
      for (int j=0;j<4;j++){
        float p = __expf(s_acc[j][r] - nm[r]);
        rs += p;
        p_lds[wave][rowl*64 + j*16 + l15] = (_Float16)p;
      }
      rs += __shfl_xor(rs,1,32); rs += __shfl_xor(rs,2,32);
      rs += __shfl_xor(rs,4,32); rs += __shfl_xor(rs,8,32);
      lrow[r] = lrow[r]*fct[r] + rs;
#pragma unroll
      for (int dt=0;dt<4;dt++) oacc[dt][r] *= fct[r];
    }
    asm volatile("s_wait_dscnt 0x0" ::: "memory");
    // ---- O += P * V ----
#pragma unroll
    for (int s2=0;s2<2;s2++){
      if (kb + 32*s2 >= NN) break;          // fully-masked half-tile
      const int ako = half ? 16 : 0;
      v8h lo = *(const v8h*)&p_lds[wave][l15*64 + 32*s2 + ako];
      v8h hi = *(const v8h*)&p_lds[wave][l15*64 + 32*s2 + ako + 8];
      v16h ap = cat16(lo, hi);
#pragma unroll
      for (int dt=0;dt<4;dt++){
        int d = dt*16 + l15;
        v16h bv = *(const v16h*)&vp[(size_t)d*VSTR + kb + 32*s2 + (half ? 16 : 0)];
        oacc[dt] = wmma_f16(ap, bv, oacc[dt]);
      }
    }
  }

  // ---- write out (B,N,H,D) ----
#pragma unroll
  for (int r=0;r<8;r++){
    int n = q0 + r + half*8;
    if (n >= NN) continue;
    float inv = 1.f / lrow[r];
#pragma unroll
    for (int dt=0;dt<4;dt++){
      int d = dt*16 + l15;
      out[(((size_t)b*NN + n)*HH + h)*DD + d] = oacc[dt][r] * inv;
    }
  }
}

// ---------------- final layernorm over C, emit f16 ----------------
__global__ __launch_bounds__(256) void ln_rows_kernel(const float* __restrict__ in,
                                                      const float* __restrict__ g,
                                                      const float* __restrict__ bb,
                                                      _Float16* __restrict__ out){
  __shared__ float red1[8];
  __shared__ float red2[8];
  const int row = blockIdx.x;
  const float* x = in + (size_t)row * CC;
  const int lane = threadIdx.x & 31, wave = threadIdx.x >> 5;

  float s = 0.f;
  for (int i = threadIdx.x; i < CC; i += 256) s += x[i];
  s = wave_sum32(s);
  if (lane == 0) red1[wave] = s;
  __syncthreads();
  float tot = 0.f;
#pragma unroll
  for (int w=0; w<8; w++) tot += red1[w];
  const float mu = tot * (1.f/CC);

  float v = 0.f;
  for (int i = threadIdx.x; i < CC; i += 256){ float d = x[i]-mu; v += d*d; }
  v = wave_sum32(v);
  if (lane == 0) red2[wave] = v;
  __syncthreads();
  float vt = 0.f;
#pragma unroll
  for (int w=0; w<8; w++) vt += red2[w];
  const float rs = rsqrtf(vt*(1.f/CC) + EPSF);

  for (int i = threadIdx.x; i < CC; i += 256)
    out[(size_t)row*CC + i] = (_Float16)((x[i]-mu)*rs*g[i] + bb[i]);
}

// ---------------- host ----------------
extern "C" void kernel_launch(void* const* d_in, const int* in_sizes, int n_in,
                              void* d_out, int out_size, void* d_ws, size_t ws_size,
                              hipStream_t stream){
  const float* x      = (const float*)d_in[0];
  const float* rope   = (const float*)d_in[1];
  const float* qkv_w  = (const float*)d_in[2];
  const float* q_bias = (const float*)d_in[3];
  const float* k_bias = (const float*)d_in[4];
  const float* v_bias = (const float*)d_in[5];
  const float* qn_g   = (const float*)d_in[6];
  const float* qn_b   = (const float*)d_in[7];
  const float* kn_g   = (const float*)d_in[8];
  const float* kn_b   = (const float*)d_in[9];
  const float* norm_g = (const float*)d_in[10];
  const float* norm_b = (const float*)d_in[11];
  const float* proj_w = (const float*)d_in[12];
  const float* proj_b = (const float*)d_in[13];
  float* out = (float*)d_out;

  char* ws = (char*)d_ws;
  size_t off = 0;
  auto alloc = [&](size_t bytes)->char*{
    char* p = ws + off;
    off += (bytes + 255) & ~(size_t)255;
    return p;
  };
  _Float16* x_h      = (_Float16*)alloc((size_t)MTOT*CC*2);
  _Float16* qkvw_h   = (_Float16*)alloc((size_t)3*CC*CC*2);
  _Float16* projw_h  = (_Float16*)alloc((size_t)CC*CC*2);
  float*    biascat  = (float*)   alloc((size_t)3*CC*4);
  float*    qkv_f32  = (float*)   alloc((size_t)MTOT*3*CC*4);
  _Float16* q_h      = (_Float16*)alloc((size_t)BB*HH*NN*DD*2);
  _Float16* k_h      = (_Float16*)alloc((size_t)BB*HH*NN*DD*2);
  _Float16* vT_h     = (_Float16*)alloc((size_t)BB*HH*DD*VSTR*2 + 4096); // + OOB-wrap pad
  float*    attn_out = (float*)   alloc((size_t)MTOT*CC*4);
  _Float16* normed_h = (_Float16*)alloc((size_t)MTOT*CC*2);

  // 1) f32 -> f16 conversions
  {
    long n1 = (long)MTOT*CC;
    f32_to_f16_kernel<<<(int)((n1+255)/256), 256, 0, stream>>>(x, x_h, n1);
    long n2 = (long)3*CC*CC;
    f32_to_f16_kernel<<<(int)((n2+255)/256), 256, 0, stream>>>(qkv_w, qkvw_h, n2);
    long n3 = (long)CC*CC;
    f32_to_f16_kernel<<<(int)((n3+255)/256), 256, 0, stream>>>(proj_w, projw_h, n3);
  }
  bias_cat_kernel<<<12, 256, 0, stream>>>(q_bias, k_bias, v_bias, biascat);

  // 2) QKV GEMM (WMMA), N = 3072 -> 12 column tiles of 256
  gemm_f16_kernel<<<dim3((MTOT+127)/128, (3*CC)/256), 256, 0, stream>>>(
      x_h, qkvw_h, biascat, qkv_f32, MTOT, 3*CC, CC);

  // 3) per-head LN + RoPE + scale + pack
  {
    long rows = (long)BB*NN*HH;
    qk_prep_kernel<<<(int)((rows+7)/8), 256, 0, stream>>>(
        qkv_f32, rope, qn_g, qn_b, kn_g, kn_b, q_h, k_h, vT_h);
  }

  // 4) flash attention (WMMA)
  flash_kernel<<<dim3((NN+63)/64, BB*HH), 128, 0, stream>>>(q_h, k_h, vT_h, attn_out);

  // 5) scale_norm
  ln_rows_kernel<<<MTOT, 256, 0, stream>>>(attn_out, norm_g, norm_b, normed_h);

  // 6) proj GEMM (WMMA) -> d_out, N = 1024 -> 4 column tiles of 256
  gemm_f16_kernel<<<dim3((MTOT+127)/128, CC/256), 256, 0, stream>>>(
      normed_h, projw_h, proj_b, out, MTOT, CC, CC);
}
